// DepthAwareTransformer_50010599194751
// MI455X (gfx1250) — compile-verified, matrix operand order checked by reading, not run on hardware
//
#include <hip/hip_runtime.h>

typedef __bf16 bf16_t;
typedef __attribute__((ext_vector_type(16))) __bf16 v16bf;
typedef __attribute__((ext_vector_type(8)))  __bf16 v8bf;
typedef __attribute__((ext_vector_type(8)))  float  v8f;
typedef __attribute__((ext_vector_type(4)))  unsigned int u32x4;
typedef __attribute__((ext_vector_type(8)))  int i32x8;
typedef __attribute__((ext_vector_type(4)))  int i32x4;

#define N_  4
#define L_  2048
#define C_  1024
#define H_  8
#define D_  128
#define M_  (N_*L_)
#define C2_ (2*C_)
#define NB_ (N_*H_)

// ---------------------------------------------------------------------------
// WMMA fragment gather (works for LDS after addrspace inference):
//   row  = lane % 16
//   lanes 0-15 : K = k0+0..7   and k0+16..23
//   lanes 16-31: K = k0+8..15  and k0+24..31
// ---------------------------------------------------------------------------
__device__ __forceinline__ v16bf load_frag(const bf16_t* p) {
  v8bf lo = *(const v8bf*)(p);
  v8bf hh = *(const v8bf*)(p + 16);
  v16bf f;
#pragma unroll
  for (int i = 0; i < 8; ++i) { f[i] = lo[i]; f[i + 8] = hh[i]; }
  return f;
}

// ---------------------------------------------------------------------------
// Tensor Data Mover: 2-D bf16 tile (tile_k x tile_rows) global -> LDS.
// D# built per CDNA5 ISA 8.3/8.4: group0 {count=1, lds_addr, global_addr,
// type=2}; group1 {data_size=1(2B), tensor dims, tile dims, dim0 stride}.
// Rows packed back-to-back in LDS (row stride = tile_k elements).
// 6-arg builtin form (amdgpu-toolchain / clang-23).
// ---------------------------------------------------------------------------
__device__ __forceinline__ void tdm_load_2d(unsigned lds_addr, const void* gaddr,
                                            int tile_k, int tile_rows, int ld)
{
  unsigned long long ga = (unsigned long long)gaddr;
  u32x4 g0;
  g0[0] = 1u;                                            // count=1 (valid)
  g0[1] = lds_addr;                                      // LDS byte address
  g0[2] = (unsigned)(ga & 0xffffffffu);                  // global_addr[31:0]
  g0[3] = (unsigned)((ga >> 32) & 0x01ffffffu) | (2u << 30); // addr[56:32]|type=2
  i32x8 g1;
  g1[0] = 1 << 16;                                       // data_size=1 (2 bytes)
  g1[1] = (tile_k & 0xffff) << 16;                       // tensor_dim0[15:0]
  g1[2] = (tile_k >> 16) | ((tile_rows & 0xffff) << 16); // dim0 hi | tensor_dim1 lo
  g1[3] = (tile_rows >> 16) | ((tile_k & 0xffff) << 16); // dim1 hi | tile_dim0
  g1[4] = (tile_rows & 0xffff);                          // tile_dim1 (tile_dim2=0)
  g1[5] = ld;                                            // tensor_dim0_stride lo
  g1[6] = 0;
  g1[7] = 0;
  i32x4 g2 = {0, 0, 0, 0};
  i32x4 g3 = {0, 0, 0, 0};
  i32x8 g4 = {0, 0, 0, 0, 0, 0, 0, 0};
  __builtin_amdgcn_tensor_load_to_lds(g0, g1, g2, g3, g4, 0);
}

// ---------------------------------------------------------------------------
// Batched bf16 GEMM:  C[M,N] = A[M,K] * Bt[N,K]^T   (both operands K-major)
// A batch offset = (z/Ha)*sA + (z%Ha)*sA2  (head batching for Q@KV GEMM)
// Block = 8 waves = 128(M) x 64(N); wave = 16x64 via 4 accumulators.
// TDM double-buffers A(128x32) and B(64x32) tiles in LDS; wave 0 issues the
// descriptors and waits TENSORcnt, barriers publish tiles to all waves.
// Requires: M covered by grid (multiple of 128 per launch), N mult of 64,
// K mult of 32.
// ---------------------------------------------------------------------------
__global__ void __launch_bounds__(256)
k_gemm(const bf16_t* __restrict__ A, const bf16_t* __restrict__ B,
       float* __restrict__ C,
       int M, int Nn, int K, int lda, int ldb, int ldc,
       long long sA, long long sA2, int Ha, long long sB, long long sC)
{
  __shared__ bf16_t smA[2][128 * 32];
  __shared__ bf16_t smB[2][64 * 32];

  int z = blockIdx.z;
  const bf16_t* Ab = A + (long long)(z / Ha) * sA + (long long)(z % Ha) * sA2;
  const bf16_t* Bb = B + (long long)z * sB;
  float*        Cb = C + (long long)z * sC;

  int lane = threadIdx.x & 31;
  int wave = threadIdx.x >> 5;
  int m0 = blockIdx.x * 128 + wave * 16;
  int n0 = blockIdx.y * 64;

  int rr = lane & 15;
  int kh = (lane >> 4) << 3;              // 0 or 8

  const char* Ag = (const char*)(Ab + (long long)blockIdx.x * 128 * lda);
  const char* Bg = (const char*)(Bb + (long long)n0 * ldb);

  bool loader = (wave == 0);
  if (loader) {
    tdm_load_2d((unsigned)(size_t)&smA[0][0], Ag, 32, 128, lda);
    tdm_load_2d((unsigned)(size_t)&smB[0][0], Bg, 32, 64, ldb);
  }

  v8f acc0 = {}, acc1 = {}, acc2 = {}, acc3 = {};
  int nbuf = 0;

  for (int k0 = 0; k0 < K; k0 += 32) {
    if (loader) {
      if (k0 + 32 < K) {                  // stream next k-tile while computing
        tdm_load_2d((unsigned)(size_t)&smA[nbuf ^ 1][0], Ag + (long long)(k0 + 32) * 2,
                    32, 128, lda);
        tdm_load_2d((unsigned)(size_t)&smB[nbuf ^ 1][0], Bg + (long long)(k0 + 32) * 2,
                    32, 64, ldb);
        __builtin_amdgcn_s_wait_tensorcnt(2);   // current pair complete
      } else {
        __builtin_amdgcn_s_wait_tensorcnt(0);
      }
    }
    __syncthreads();                      // publish LDS tiles to all waves

    const bf16_t* ab  = &smA[nbuf][(wave * 16 + rr) * 32 + kh];
    const bf16_t* bb0 = &smB[nbuf][rr * 32 + kh];
    v16bf a  = load_frag(ab);
    v16bf b0 = load_frag(bb0);
    v16bf b1 = load_frag(bb0 + 16 * 32);
    v16bf b2 = load_frag(bb0 + 32 * 32);
    v16bf b3 = load_frag(bb0 + 48 * 32);
    acc0 = __builtin_amdgcn_wmma_f32_16x16x32_bf16(false, a, false, b0, (short)0, acc0, false, false);
    acc1 = __builtin_amdgcn_wmma_f32_16x16x32_bf16(false, a, false, b1, (short)0, acc1, false, false);
    acc2 = __builtin_amdgcn_wmma_f32_16x16x32_bf16(false, a, false, b2, (short)0, acc2, false, false);
    acc3 = __builtin_amdgcn_wmma_f32_16x16x32_bf16(false, a, false, b3, (short)0, acc3, false, false);

    __syncthreads();                      // all reads done before TDM overwrites
    nbuf ^= 1;
  }

  // C/D layout: VGPR r, lanes 0-15 -> M=r, N=lane; lanes 16-31 -> M=r+8.
  int row0 = m0 + ((lane >> 4) << 3);
  int col  = n0 + rr;
#pragma unroll
  for (int r = 0; r < 8; ++r) {
    long long off = (long long)(row0 + r) * ldc;
    Cb[off + col     ] = acc0[r];
    Cb[off + col + 16] = acc1[r];
    Cb[off + col + 32] = acc2[r];
    Cb[off + col + 48] = acc3[r];
  }
}

// ---------------------------------------------------------------------------
// Tiled transpose f32 -> bf16 with optional elementwise op.
// dst[c*ld_dst + r] = op(src[r*ld_src + c]) ; batch z: src += (z/Ha)*sN+(z%Ha)*sH
// mode: 0=copy, 1=elu(x)+1, 2=x*scale
// ---------------------------------------------------------------------------
__global__ void __launch_bounds__(256)
k_transpose(const float* __restrict__ src, bf16_t* __restrict__ dst,
            int R, int Cc, int ld_src, int ld_dst,
            int Ha, long long sN, long long sH, long long sD,
            int mode, float scale)
{
  __shared__ float tile[32][33];
  int z = blockIdx.z;
  const float* s = src + (long long)(z / Ha) * sN + (long long)(z % Ha) * sH;
  bf16_t*      d = dst + (long long)z * sD;
  int r0 = blockIdx.x * 32;
  int c0 = blockIdx.y * 32;
  for (int i = threadIdx.y; i < 32; i += 8) {
    int r = r0 + i, c = c0 + threadIdx.x;
    float v = 0.f;
    if (r < R && c < Cc) v = s[(long long)r * ld_src + c];
    if (mode == 1)      v = (v > 0.f) ? (v + 1.f) : __expf(v);
    else if (mode == 2) v = v * scale;
    tile[i][threadIdx.x] = v;
  }
  __syncthreads();
  for (int i = threadIdx.y; i < 32; i += 8) {
    int c = c0 + i, r = r0 + threadIdx.x;
    if (r < R && c < Cc) d[(long long)c * ld_dst + r] = (bf16_t)tile[threadIdx.x][i];
  }
}

// elementwise f32 -> bf16; mode: 0=copy, 1=relu, 2=elu(x)+1
__global__ void k_cvt(const float* __restrict__ src, bf16_t* __restrict__ dst,
                      long long n, int mode)
{
  long long i = (long long)blockIdx.x * blockDim.x + threadIdx.x;
  if (i >= n) return;
  float v = src[i];
  if (mode == 1)      v = fmaxf(v, 0.f);
  else if (mode == 2) v = (v > 0.f) ? (v + 1.f) : __expf(v);
  dst[i] = (bf16_t)v;
}

// out_f32 = a (+ b), out_bf16 = bf16(out_f32)
__global__ void k_addcvt(const float* __restrict__ a, const float* __restrict__ b,
                         float* __restrict__ of, bf16_t* __restrict__ ob, long long n)
{
  long long i = (long long)blockIdx.x * blockDim.x + threadIdx.x;
  if (i >= n) return;
  float v = a[i] + (b ? b[i] : 0.f);
  of[i] = v;
  ob[i] = (bf16_t)v;
}

// Ksum[row] = sum_l Kt[row*L + l]   (one 256-thread block per row)
__global__ void __launch_bounds__(256)
k_rowsum(const bf16_t* __restrict__ Kt, float* __restrict__ Ksum, int Ll)
{
  __shared__ float sd[256];
  long long row = blockIdx.x;
  const bf16_t* p = Kt + row * Ll;
  float s = 0.f;
  for (int i = threadIdx.x; i < Ll; i += 256) s += (float)p[i];
  sd[threadIdx.x] = s; __syncthreads();
  for (int st = 128; st > 0; st >>= 1) {
    if ((int)threadIdx.x < st) sd[threadIdx.x] += sd[threadIdx.x + st];
    __syncthreads();
  }
  if (threadIdx.x == 0) Ksum[row] = sd[0];
}

// Z = 1/(Q'.Ksum + eps); msg[n,l,h*D+d] = attn[n,h,l,d] * Z * vlen
// grid.x = N*L*H, block = D (=128)
__global__ void __launch_bounds__(128)
k_zmerge(const bf16_t* __restrict__ Qp, const float* __restrict__ Ksum,
         const float* __restrict__ attn, bf16_t* __restrict__ msg,
         int Hh, int Ll, int Cc, int Dd, float vlen)
{
  __shared__ float sd[128];
  int idx = blockIdx.x;
  int h = idx % Hh;
  int l = (idx / Hh) % Ll;
  int n = idx / (Hh * Ll);
  int d = threadIdx.x;
  long long qoff = ((long long)(n * Ll + l)) * Cc + (long long)h * Dd;
  long long koff = (long long)(n * Hh + h) * Dd;
  sd[d] = (float)Qp[qoff + d] * Ksum[koff + d];
  __syncthreads();
  for (int st = 64; st > 0; st >>= 1) {
    if (d < st) sd[d] += sd[d + st];
    __syncthreads();
  }
  float Z = 1.f / (sd[0] + 1e-6f);
  long long aoff = (((long long)(n * Hh + h)) * Ll + l) * Dd;
  msg[qoff + d] = (bf16_t)(attn[aoff + d] * Z * vlen);
}

// out = res + LayerNorm(y)*g + b  (one 256-thread block per row of C)
__global__ void __launch_bounds__(256)
k_ln_res(const float* __restrict__ res, const float* __restrict__ y,
         const float* __restrict__ g, const float* __restrict__ bb,
         float* __restrict__ of, bf16_t* __restrict__ ob, int Cc)
{
  __shared__ float sd[256];
  __shared__ float sm, sv;
  long long r = blockIdx.x;
  const float* yr = y + r * Cc;
  int t = threadIdx.x;
  float s = 0.f;
  for (int i = t; i < Cc; i += 256) s += yr[i];
  sd[t] = s; __syncthreads();
  for (int st = 128; st > 0; st >>= 1) { if (t < st) sd[t] += sd[t + st]; __syncthreads(); }
  if (t == 0) sm = sd[0] / (float)Cc;
  __syncthreads();
  float m = sm;
  s = 0.f;
  for (int i = t; i < Cc; i += 256) { float dv = yr[i] - m; s += dv * dv; }
  sd[t] = s; __syncthreads();
  for (int st = 128; st > 0; st >>= 1) { if (t < st) sd[t] += sd[t + st]; __syncthreads(); }
  if (t == 0) sv = rsqrtf(sd[0] / (float)Cc + 1e-5f);
  __syncthreads();
  float rs = sv;
  for (int i = t; i < Cc; i += 256) {
    float v = res[r * Cc + i] + (yr[i] - m) * rs * g[i] + bb[i];
    of[r * Cc + i] = v;
    if (ob) ob[r * Cc + i] = (bf16_t)v;
  }
}

// ---------------------------------------------------------------------------
extern "C" void kernel_launch(void* const* d_in, const int* in_sizes, int n_in,
                              void* d_out, int out_size, void* d_ws, size_t ws_size,
                              hipStream_t stream)
{
  (void)in_sizes; (void)n_in; (void)out_size; (void)ws_size;

  const float* depth_feat   = (const float*)d_in[0];
  const float* context_feat = (const float*)d_in[1];
  const float* depth_pos    = (const float*)d_in[2];
  const float* e_g1 = (const float*)d_in[9];
  const float* e_b1 = (const float*)d_in[10];
  const float* e_g2 = (const float*)d_in[11];
  const float* e_b2 = (const float*)d_in[12];
  const float* d_g0 = (const float*)d_in[23];
  const float* d_b0 = (const float*)d_in[24];
  const float* d_g1 = (const float*)d_in[25];
  const float* d_b1 = (const float*)d_in[26];
  const float* d_g2 = (const float*)d_in[27];
  const float* d_b2 = (const float*)d_in[28];

  char* w = (char*)d_ws;
  size_t off = 0;
  auto alloc = [&](size_t bytes) -> void* {
    void* p = w + off;
    off = (off + bytes + 255) & ~(size_t)255;
    return p;
  };

  // bf16 transposed weights [Nout, K]
  bf16_t* w_ewq  = (bf16_t*)alloc((size_t)C_  * C_  * 2);
  bf16_t* w_ewk  = (bf16_t*)alloc((size_t)C_  * C_  * 2);
  bf16_t* w_ewv  = (bf16_t*)alloc((size_t)C_  * C_  * 2);
  bf16_t* w_ewm  = (bf16_t*)alloc((size_t)C_  * C_  * 2);
  bf16_t* w_ew1  = (bf16_t*)alloc((size_t)C_  * C2_ * 2);
  bf16_t* w_ew2  = (bf16_t*)alloc((size_t)C2_ * C_  * 2);
  bf16_t* w_dwq0 = (bf16_t*)alloc((size_t)C_  * C_  * 2);
  bf16_t* w_dwk0 = (bf16_t*)alloc((size_t)C_  * C_  * 2);
  bf16_t* w_dwv0 = (bf16_t*)alloc((size_t)C_  * C_  * 2);
  bf16_t* w_dwm0 = (bf16_t*)alloc((size_t)C_  * C_  * 2);
  bf16_t* w_dwq1 = (bf16_t*)alloc((size_t)C_  * C_  * 2);
  bf16_t* w_dwk1 = (bf16_t*)alloc((size_t)C_  * C_  * 2);
  bf16_t* w_dwv1 = (bf16_t*)alloc((size_t)C_  * C_  * 2);
  bf16_t* w_dwm1 = (bf16_t*)alloc((size_t)C_  * C_  * 2);
  bf16_t* w_dw1  = (bf16_t*)alloc((size_t)C_  * C2_ * 2);
  bf16_t* w_dw2  = (bf16_t*)alloc((size_t)C2_ * C_  * 2);

  // activations
  float*  ctx_f  = (float*) alloc((size_t)M_ * C_  * 4);
  bf16_t* ctx_b  = (bf16_t*)alloc((size_t)M_ * C_  * 2);
  float*  x_f    = (float*) alloc((size_t)M_ * C_  * 4);
  bf16_t* x_b    = (bf16_t*)alloc((size_t)M_ * C_  * 2);
  float*  q_f    = (float*) alloc((size_t)M_ * C_  * 4);  // also msg@wm output
  float*  k_f    = (float*) alloc((size_t)M_ * C_  * 4);  // also attn out & ffn out
  float*  v_f    = (float*) alloc((size_t)M_ * C_  * 4);
  bf16_t* qp_b   = (bf16_t*)alloc((size_t)M_ * C_  * 2);
  bf16_t* kt_b   = (bf16_t*)alloc((size_t)NB_ * D_ * L_ * 2);
  bf16_t* vt_b   = (bf16_t*)alloc((size_t)NB_ * D_ * L_ * 2);
  float*  ksum   = (float*) alloc((size_t)NB_ * D_ * 4);
  float*  kvt_f  = (float*) alloc((size_t)NB_ * D_ * D_ * 4);
  bf16_t* kvt_b  = (bf16_t*)alloc((size_t)NB_ * D_ * D_ * 2);
  bf16_t* msg_b  = (bf16_t*)alloc((size_t)M_ * C_  * 2);
  float*  ffn_f  = (float*) alloc((size_t)M_ * C2_ * 4);
  bf16_t* ffn_b  = (bf16_t*)alloc((size_t)M_ * C2_ * 2);

  auto gemm = [&](const bf16_t* A, const bf16_t* B, float* Cp,
                  int M, int Nn, int K, int lda, int ldb, int ldc,
                  long long sA, long long sA2, int Ha,
                  long long sB, long long sC, int batch) {
    dim3 g((M + 127) / 128, Nn / 64, batch);
    k_gemm<<<g, dim3(256, 1, 1), 0, stream>>>(A, B, Cp, M, Nn, K, lda, ldb, ldc,
                                              sA, sA2, Ha, sB, sC);
  };
  auto trw = [&](const float* src, bf16_t* dst, int K, int Nn) {
    dim3 g(K / 32, Nn / 32, 1);
    k_transpose<<<g, dim3(32, 8), 0, stream>>>(src, dst, K, Nn, Nn, K,
                                               1, 0, 0, 0, 0, 0.f);
  };
  auto cvt = [&](const float* s, bf16_t* d, long long n, int mode) {
    k_cvt<<<dim3((unsigned)((n + 255) / 256)), 256, 0, stream>>>(s, d, n, mode);
  };
  auto lnres = [&](const float* res, const float* y, const float* g, const float* b,
                   float* of, bf16_t* ob) {
    k_ln_res<<<M_, 256, 0, stream>>>(res, y, g, b, of, ob, C_);
  };

  // Linear attention block: msg_out = Wm-projected kernelized attention
  auto attention = [&](const bf16_t* xq, const bf16_t* xkv,
                       const bf16_t* wq, const bf16_t* wk, const bf16_t* wv,
                       const bf16_t* wm, float* msg_out) {
    gemm(xq,  wq, q_f, M_, C_, C_, C_, C_, C_, 0, 0, 1, 0, 0, 1);
    gemm(xkv, wk, k_f, M_, C_, C_, C_, C_, C_, 0, 0, 1, 0, 0, 1);
    gemm(xkv, wv, v_f, M_, C_, C_, C_, C_, C_, 0, 0, 1, 0, 0, 1);
    cvt(q_f, qp_b, (long long)M_ * C_, 2);               // Q' = elu(q)+1
    {   // K' transposed to [n,h,d,l] with elu+1
      dim3 g(L_ / 32, D_ / 32, NB_);
      k_transpose<<<g, dim3(32, 8), 0, stream>>>(k_f, kt_b, L_, D_, C_, L_,
          H_, (long long)L_ * C_, (long long)D_, (long long)D_ * L_, 1, 0.f);
    }
    {   // V/L transposed to [n,h,dv,l]
      dim3 g(L_ / 32, D_ / 32, NB_);
      k_transpose<<<g, dim3(32, 8), 0, stream>>>(v_f, vt_b, L_, D_, C_, L_,
          H_, (long long)L_ * C_, (long long)D_, (long long)D_ * L_, 2, 1.0f / L_);
    }
    k_rowsum<<<NB_ * D_, 256, 0, stream>>>(kt_b, ksum, L_);
    // KVt[dv,d] = sum_l Vt[dv,l]*Kt[d,l]   (batched over n,h)
    gemm(vt_b, kt_b, kvt_f, D_, D_, L_, L_, L_, D_,
         (long long)D_ * L_, 0, 1, (long long)D_ * L_, (long long)D_ * D_, NB_);
    cvt(kvt_f, kvt_b, (long long)NB_ * D_ * D_, 0);
    // attn[l,dv] = sum_d Qp[l,d]*KVt[dv,d]  (A uses head offsets into [M,C])
    gemm(qp_b, kvt_b, k_f, L_, D_, D_, C_, D_, D_,
         (long long)L_ * C_, (long long)D_, H_,
         (long long)D_ * D_, (long long)L_ * D_, NB_);
    k_zmerge<<<N_ * L_ * H_, D_, 0, stream>>>(qp_b, ksum, k_f, msg_b,
                                              H_, L_, C_, D_, (float)L_);
    gemm(msg_b, wm, msg_out, M_, C_, C_, C_, C_, C_, 0, 0, 1, 0, 0, 1);
  };

  auto ffn = [&](const bf16_t* xb, const bf16_t* w1t, const bf16_t* w2t, float* out) {
    gemm(xb, w1t, ffn_f, M_, C2_, C_, C_, C_, C2_, 0, 0, 1, 0, 0, 1);
    cvt(ffn_f, ffn_b, (long long)M_ * C2_, 1);           // relu
    gemm(ffn_b, w2t, out, M_, C_, C2_, C2_, C2_, C_, 0, 0, 1, 0, 0, 1);
  };

  // ---- weight prep (transpose + bf16) ----
  trw((const float*)d_in[3],  w_ewq,  C_,  C_);
  trw((const float*)d_in[4],  w_ewk,  C_,  C_);
  trw((const float*)d_in[5],  w_ewv,  C_,  C_);
  trw((const float*)d_in[6],  w_ewm,  C_,  C_);
  trw((const float*)d_in[7],  w_ew1,  C_,  C2_);
  trw((const float*)d_in[8],  w_ew2,  C2_, C_);
  trw((const float*)d_in[13], w_dwq0, C_,  C_);
  trw((const float*)d_in[14], w_dwk0, C_,  C_);
  trw((const float*)d_in[15], w_dwv0, C_,  C_);
  trw((const float*)d_in[16], w_dwm0, C_,  C_);
  trw((const float*)d_in[17], w_dwq1, C_,  C_);
  trw((const float*)d_in[18], w_dwk1, C_,  C_);
  trw((const float*)d_in[19], w_dwv1, C_,  C_);
  trw((const float*)d_in[20], w_dwm1, C_,  C_);
  trw((const float*)d_in[21], w_dw1,  C_,  C2_);
  trw((const float*)d_in[22], w_dw2,  C2_, C_);

  const long long MC = (long long)M_ * C_;

  // ---- encoder ----
  k_addcvt<<<(unsigned)((MC + 255) / 256), 256, 0, stream>>>(
      context_feat, depth_pos, ctx_f, ctx_b, MC);
  attention(ctx_b, ctx_b, w_ewq, w_ewk, w_ewv, w_ewm, q_f);
  lnres(ctx_f, q_f, e_g1, e_b1, x_f, x_b);               // x1 = ctx + LN(msg)
  ffn(x_b, w_ew1, w_ew2, k_f);
  lnres(x_f, k_f, e_g2, e_b2, ctx_f, ctx_b);             // ctx = x1 + LN(ffn)

  // ---- decoder ----
  k_addcvt<<<(unsigned)((MC + 255) / 256), 256, 0, stream>>>(
      depth_feat, nullptr, x_f, x_b, MC);
  attention(x_b, x_b, w_dwq0, w_dwk0, w_dwv0, w_dwm0, q_f);
  lnres(x_f, q_f, d_g0, d_b0, x_f, x_b);
  attention(x_b, ctx_b, w_dwq1, w_dwk1, w_dwv1, w_dwm1, q_f);
  lnres(x_f, q_f, d_g1, d_b1, x_f, x_b);
  ffn(x_b, w_dw1, w_dw2, k_f);
  lnres(x_f, k_f, d_g2, d_b2, (float*)d_out, nullptr);
}
